// MultiHeadAttention_45853070852704
// MI455X (gfx1250) — matrix operand order repaired and checked by reading, not compiled, not run on hardware
//
#include <hip/hip_runtime.h>
#include <hip/hip_bf16.h>

// ---------------------------------------------------------------------------
// MHA forward for MI455X (gfx1250, wave32, WMMA + async global->LDS copies).
//   B=4, TQ=TK=2048, D=512, H=8, Dh=64
// Pipeline:
//   1) cvt4_kernel:  Wq/Wk/Wv/Wo fp32 -> bf16 (workspace)
//   2) gemm_xwT<fp32 in, headsplit bf16 out> x3 : Q/K/V projections
//   3) attn_kernel:  two-pass flash-style causal attention, 64-wide K tiles,
//                    writes w (fp32) exactly once, bf16 head-merged output
//   4) gemm_xwT<bf16 in, plain fp32 out>: output projection
// All GEMMs: v_wmma_f32_16x16x32_bf16. All bf16 tile staging:
// global_load_async_to_lds_b128 (ASYNCcnt) when the builtin is available.
// ---------------------------------------------------------------------------

#define CB 4
#define CTQ 2048
#define CTK 2048
#define CD 512
#define CH 8
#define CDH 64

typedef __bf16 bf16_t;
typedef __attribute__((ext_vector_type(16))) __bf16 v16bf;
typedef __attribute__((ext_vector_type(8)))  float  v8f;
typedef __attribute__((ext_vector_type(4)))  int    v4i;

#if defined(__has_builtin)
# if __has_builtin(__builtin_amdgcn_global_load_async_to_lds_b128)
#  define HAVE_ASYNC_LDS 1
# endif
#endif
#ifndef HAVE_ASYNC_LDS
# define HAVE_ASYNC_LDS 0
#endif

#define AS1 __attribute__((address_space(1)))
#define AS3 __attribute__((address_space(3)))

__device__ __forceinline__ void async_wait0() {
#if HAVE_ASYNC_LDS
# if __has_builtin(__builtin_amdgcn_s_wait_asynccnt)
  __builtin_amdgcn_s_wait_asynccnt(0);
# else
  asm volatile("s_wait_asynccnt 0x0" ::: "memory");
# endif
#endif
}

// Copy a ROWS x COLS_E bf16 tile (row-major, 16B-aligned rows) from global to
// LDS with row stride LSTRIDE_E (elements; LSTRIDE_E*2 must be 16B multiple).
template <int ROWS, int COLS_E, int LSTRIDE_E>
__device__ __forceinline__ void async_tile_load(const bf16_t* __restrict__ g,
                                                int gstride_e,
                                                bf16_t* __restrict__ lds) {
  constexpr int CPR = (COLS_E * 2) / 16;   // 16B chunks per row
  constexpr int TOT = ROWS * CPR;
  static_assert((LSTRIDE_E * 2) % 16 == 0, "lds stride must be 16B aligned");
#pragma unroll
  for (int i = threadIdx.x; i < TOT; i += 128) {
    const int r = i / CPR, c = i - r * CPR;
#if HAVE_ASYNC_LDS
    __builtin_amdgcn_global_load_async_to_lds_b128(
        (AS1 v4i*)(g + (size_t)r * gstride_e + c * 8),
        (AS3 v4i*)(lds + r * LSTRIDE_E + c * 8), 0, 0);
#else
    *(int4*)(lds + r * LSTRIDE_E + c * 8) =
        *(const int4*)(g + (size_t)r * gstride_e + c * 8);
#endif
  }
}

__device__ __forceinline__ v8f wmma_bf16(v16bf a, v16bf b, v8f c) {
  return __builtin_amdgcn_wmma_f32_16x16x32_bf16(false, a, false, b,
                                                 (short)0, c, false, false);
}

// A fragment: 16x32 bf16 tile, lds row-major [m][k], stride ldk.
__device__ __forceinline__ v16bf load_frag_a(const bf16_t* lds, int ldk) {
  const int lane = threadIdx.x & 31;
  const int row  = lane & 15;
  const int kb   = (lane >> 4) << 4;   // 0 or 16
  union { v16bf v; bf16_t e[16]; } u;
#pragma unroll
  for (int i = 0; i < 8; ++i) u.e[i]     = lds[row * ldk + kb + i];
#pragma unroll
  for (int i = 0; i < 8; ++i) u.e[8 + i] = lds[row * ldk + kb + 8 + i];
  return u.v;
}

// B fragment (K x N), lds row-major [k][n], stride ldn.
__device__ __forceinline__ v16bf load_frag_b(const bf16_t* lds, int ldn) {
  const int lane = threadIdx.x & 31;
  const int n  = lane & 15;
  const int kb = (lane >> 4) << 4;     // 0 or 16
  union { v16bf v; bf16_t e[16]; } u;
#pragma unroll
  for (int i = 0; i < 16; ++i) u.e[i] = lds[(kb + i) * ldn + n];
  return u.v;
}

// B fragment when lds holds the matrix transposed: [n][k] layout, k contiguous.
__device__ __forceinline__ v16bf load_frag_b_nk(const bf16_t* lds, int ldk) {
  const int lane = threadIdx.x & 31;
  const int n  = lane & 15;
  const int kb = (lane >> 4) << 4;
  union { v16bf v; bf16_t e[16]; } u;
#pragma unroll
  for (int i = 0; i < 16; ++i) u.e[i] = lds[n * ldk + kb + i];
  return u.v;
}

// ---------------------------------------------------------------------------
__global__ __launch_bounds__(256)
void cvt4_kernel(const float* w0, const float* w1, const float* w2, const float* w3,
                 bf16_t* o0, bf16_t* o1, bf16_t* o2, bf16_t* o3) {
  int i = blockIdx.x * 256 + threadIdx.x;
  if (i < CD * CD) {
    o0[i] = (bf16_t)w0[i];
    o1[i] = (bf16_t)w1[i];
    o2[i] = (bf16_t)w2[i];
    o3[i] = (bf16_t)w3[i];
  }
}

// ---------------------------------------------------------------------------
// Y = X @ W^T + bias.  X: [M, K(=D)] (fp32 or bf16), W: [N(=D), K] bf16.
// Block: 128 threads (4 waves), block tile 64(M) x 64(N), k-step 32.
// W tile is async-copied verbatim ([n][k]); the b_nk fragment gather does the
// transpose at ds-load time.
// ---------------------------------------------------------------------------
template <bool IN_BF16, bool HEADSPLIT>
__global__ __launch_bounds__(128)
void gemm_xwT(const void* __restrict__ Xv, const bf16_t* __restrict__ W,
              const float* __restrict__ bias, void* __restrict__ Outv) {
  __shared__ bf16_t As[64][40];   // [m][k], stride 80B
  __shared__ bf16_t Bs[64][40];   // [n][k] = raw W rows

  const int wave = threadIdx.x >> 5;
  const int lane = threadIdx.x & 31;
  const int m0 = blockIdx.y * 64;
  const int n0 = blockIdx.x * 64;

  v8f acc[4] = {};

  for (int k0 = 0; k0 < CD; k0 += 32) {
    if (IN_BF16) {
      async_tile_load<64, 32, 40>((const bf16_t*)Xv + (size_t)m0 * CD + k0, CD,
                                  &As[0][0]);
    } else {
      for (int i = threadIdx.x; i < 64 * 32; i += 128) {
        const int r = i >> 5, c = i & 31;
        As[r][c] = (bf16_t)((const float*)Xv)[(size_t)(m0 + r) * CD + k0 + c];
      }
    }
    async_tile_load<64, 32, 40>(W + (size_t)n0 * CD + k0, CD, &Bs[0][0]);
    async_wait0();
    __syncthreads();

    const v16bf a = load_frag_a(&As[wave * 16][0], 40);
#pragma unroll
    for (int j = 0; j < 4; ++j) {
      const v16bf b = load_frag_b_nk(&Bs[j * 16][0], 40);
      acc[j] = wmma_bf16(a, b, acc[j]);
    }
    __syncthreads();
  }

  const int col = lane & 15;
  const int rbase = (lane >> 4) << 3;
#pragma unroll
  for (int j = 0; j < 4; ++j) {
    const int n = n0 + j * 16 + col;
    const float bv = bias[n];
#pragma unroll
    for (int r = 0; r < 8; ++r) {
      const int m = m0 + wave * 16 + rbase + r;
      const float val = acc[j][r] + bv;
      if (HEADSPLIT) {
        const int bb = m / CTQ, t = m % CTQ;
        const int h = n >> 6, dh = n & 63;
        ((bf16_t*)Outv)[((((size_t)bb * CH + h) * CTQ) + t) * CDH + dh] = (bf16_t)val;
      } else {
        ((float*)Outv)[(size_t)m * CD + n] = val;
      }
    }
  }
}

// ---------------------------------------------------------------------------
// Two-pass causal attention per (b, h, 64-row q tile). 4 waves, 64-wide K tiles.
// ---------------------------------------------------------------------------
__global__ __launch_bounds__(128)
void attn_kernel(const bf16_t* __restrict__ Qh, const bf16_t* __restrict__ Kh,
                 const bf16_t* __restrict__ Vh, float* __restrict__ Wout,
                 bf16_t* __restrict__ Xattn) {
  const int q0 = blockIdx.x * 64;
  const int h  = blockIdx.y;
  const int b  = blockIdx.z;

  const bf16_t* Q = Qh + ((size_t)b * CH + h) * CTQ * CDH;
  const bf16_t* K = Kh + ((size_t)b * CH + h) * CTK * CDH;
  const bf16_t* V = Vh + ((size_t)b * CH + h) * CTK * CDH;
  float* Wrow = Wout + ((size_t)b * CH + h) * (size_t)CTQ * CTK;

  __shared__ bf16_t Qs[64][72];
  __shared__ bf16_t Ks[64][72];
  __shared__ bf16_t Vs[64][72];
  __shared__ bf16_t Ps[64][72];

  const int wave = threadIdx.x >> 5;
  const int lane = threadIdx.x & 31;
  const int col  = lane & 15;
  const int qrow_base = wave * 16 + ((lane >> 4) << 3);
  const float scale = 0.125f;  // 1/sqrt(64)
  const int ktend = q0 + 64;   // causal: only j <= q0+63 can be live

  async_tile_load<64, 64, 72>(Q + (size_t)q0 * CDH, CDH, &Qs[0][0]);
  async_wait0();
  __syncthreads();

  float mrow[8], lrow[8];
#pragma unroll
  for (int r = 0; r < 8; ++r) { mrow[r] = -3.0e38f; lrow[r] = 0.f; }

  // ---------------- PASS 1: row max / row sum (online) ----------------
  for (int j0 = 0; j0 < ktend; j0 += 64) {
    async_tile_load<64, 64, 72>(K + (size_t)j0 * CDH, CDH, &Ks[0][0]);
    async_wait0();
    __syncthreads();

    v8f s[4] = {};
#pragma unroll
    for (int ksb = 0; ksb < 2; ++ksb) {
      const v16bf a = load_frag_a(&Qs[wave * 16][ksb * 32], 72);
#pragma unroll
      for (int jt = 0; jt < 4; ++jt) {
        const v16bf bb = load_frag_b_nk(&Ks[jt * 16][ksb * 32], 72);
        s[jt] = wmma_bf16(a, bb, s[jt]);
      }
    }
    __syncthreads();

#pragma unroll
    for (int r = 0; r < 8; ++r) {
      const int q = q0 + qrow_base + r;
      float sv[4];
      float best = -3.0e38f;
#pragma unroll
      for (int jt = 0; jt < 4; ++jt) {
        const int j = j0 + jt * 16 + col;
        float vv = s[jt][r] * scale;
        vv = (j <= q) ? vv : -3.0e38f;
        sv[jt] = vv;
        best = fmaxf(best, vv);
      }
#pragma unroll
      for (int m = 1; m < 16; m <<= 1) best = fmaxf(best, __shfl_xor(best, m, 32));
      const float mnew = fmaxf(mrow[r], best);
      float psum = 0.f;
#pragma unroll
      for (int jt = 0; jt < 4; ++jt) psum += __expf(sv[jt] - mnew);
#pragma unroll
      for (int m = 1; m < 16; m <<= 1) psum += __shfl_xor(psum, m, 32);
      lrow[r] = lrow[r] * __expf(mrow[r] - mnew) + psum;
      mrow[r] = mnew;
    }
  }

  float linv[8];
#pragma unroll
  for (int r = 0; r < 8; ++r) linv[r] = 1.0f / lrow[r];

  // ---------------- PASS 2: write w, accumulate O = P @ V ----------------
  v8f o[4] = {};
  for (int j0 = 0; j0 < ktend; j0 += 64) {
    async_tile_load<64, 64, 72>(K + (size_t)j0 * CDH, CDH, &Ks[0][0]);
    async_tile_load<64, 64, 72>(V + (size_t)j0 * CDH, CDH, &Vs[0][0]);
    async_wait0();
    __syncthreads();

    v8f s[4] = {};
#pragma unroll
    for (int ksb = 0; ksb < 2; ++ksb) {
      const v16bf a = load_frag_a(&Qs[wave * 16][ksb * 32], 72);
#pragma unroll
      for (int jt = 0; jt < 4; ++jt) {
        const v16bf bb = load_frag_b_nk(&Ks[jt * 16][ksb * 32], 72);
        s[jt] = wmma_bf16(a, bb, s[jt]);
      }
    }

#pragma unroll
    for (int r = 0; r < 8; ++r) {
      const int lq = qrow_base + r;
      const int q = q0 + lq;
#pragma unroll
      for (int jt = 0; jt < 4; ++jt) {
        const int jl = jt * 16 + col;
        const int j = j0 + jl;
        const float vv = s[jt][r] * scale;
        const float p = (j <= q) ? __expf(vv - mrow[r]) * linv[r] : 0.f;
        Wrow[(size_t)q * CTK + j] = p;           // normalized prob, written once
        Ps[lq][jl] = (bf16_t)p;                  // wave-local rows only
      }
    }

    // O += P @ V  (A = Ps 16x32 chunks, B = Vs rows, same-wave LDS ordering)
#pragma unroll
    for (int ksb = 0; ksb < 2; ++ksb) {
      const v16bf a = load_frag_a(&Ps[wave * 16][ksb * 32], 72);
#pragma unroll
      for (int nt = 0; nt < 4; ++nt) {
        const v16bf bb = load_frag_b(&Vs[ksb * 32][nt * 16], 72);
        o[nt] = wmma_bf16(a, bb, o[nt]);
      }
    }
    __syncthreads();
  }

  // zero the fully-masked columns j >= ktend (output buffer is poisoned)
  for (int j0 = ktend; j0 < CTK; j0 += 64) {
#pragma unroll
    for (int r = 0; r < 8; ++r) {
      const int q = q0 + qrow_base + r;
#pragma unroll
      for (int jt = 0; jt < 4; ++jt)
        Wrow[(size_t)q * CTK + j0 + jt * 16 + col] = 0.f;
    }
  }

  // store O head-merged: Xattn[b, t, h*64 + dh] (bf16)
#pragma unroll
  for (int nt = 0; nt < 4; ++nt) {
    const int dh = nt * 16 + col;
#pragma unroll
    for (int r = 0; r < 8; ++r) {
      const int t = q0 + qrow_base + r;
      Xattn[((size_t)b * CTQ + t) * CD + h * CDH + dh] = (bf16_t)o[nt][r];
    }
  }
}

// ---------------------------------------------------------------------------
extern "C" void kernel_launch(void* const* d_in, const int* in_sizes, int n_in,
                              void* d_out, int out_size, void* d_ws, size_t ws_size,
                              hipStream_t stream) {
  (void)in_sizes; (void)n_in; (void)out_size; (void)ws_size;

  const float* q  = (const float*)d_in[0];
  const float* k  = (const float*)d_in[1];
  const float* v  = (const float*)d_in[2];
  const float* Wq = (const float*)d_in[3];
  const float* bq = (const float*)d_in[4];
  const float* Wk = (const float*)d_in[5];
  const float* bk = (const float*)d_in[6];
  const float* Wv = (const float*)d_in[7];
  const float* bv = (const float*)d_in[8];
  const float* Wo = (const float*)d_in[9];
  const float* bo = (const float*)d_in[10];

  bf16_t* Wq_b = (bf16_t*)d_ws;
  bf16_t* Wk_b = Wq_b + (size_t)CD * CD;
  bf16_t* Wv_b = Wk_b + (size_t)CD * CD;
  bf16_t* Wo_b = Wv_b + (size_t)CD * CD;
  bf16_t* Qh   = Wo_b + (size_t)CD * CD;
  bf16_t* Kh   = Qh + (size_t)CB * CH * CTQ * CDH;
  bf16_t* Vh   = Kh + (size_t)CB * CH * CTK * CDH;
  bf16_t* Xa   = Vh + (size_t)CB * CH * CTK * CDH;

  float* out   = (float*)d_out;
  float* w_out = out + (size_t)CB * CTQ * CD;

  cvt4_kernel<<<(CD * CD + 255) / 256, 256, 0, stream>>>(
      Wq, Wk, Wv, Wo, Wq_b, Wk_b, Wv_b, Wo_b);

  const dim3 ggrid(CD / 64, (CB * CTQ) / 64);
  gemm_xwT<false, true><<<ggrid, 128, 0, stream>>>(q, Wq_b, bq, Qh);
  gemm_xwT<false, true><<<ggrid, 128, 0, stream>>>(k, Wk_b, bk, Kh);
  gemm_xwT<false, true><<<ggrid, 128, 0, stream>>>(v, Wv_b, bv, Vh);

  attn_kernel<<<dim3(CTQ / 64, CH, CB), 128, 0, stream>>>(Qh, Kh, Vh, w_out, Xa);

  gemm_xwT<true, false><<<ggrid, 128, 0, stream>>>(Xa, Wo_b, bo, out);
}
